// SparseMOE_4930622456423
// MI455X (gfx1250) — matrix operand order repaired and checked by reading, not compiled, not run on hardware
//
#include <hip/hip_runtime.h>
#include <hip/hip_bf16.h>
#include <math.h>

// Problem constants (from reference): B=4, S=4096, H=1024, E=8, K=2
#define TOK   16384          // B*S
#define HSZ   1024
#define NEXP  8
#define TOPK  2

// GEMM tiling
#define MT    128            // tokens per block tile
#define NT    128            // output columns per block tile
#define KT    32             // K-step (matches wmma 16x16x32 bf16)
#define ASTR  40             // A LDS row stride in bf16 elems (32 + 8 pad)
#define BSTR  136            // B LDS row stride in bf16 elems (128 + 8 pad)

typedef __attribute__((ext_vector_type(16))) __bf16 v16bf;
typedef __attribute__((ext_vector_type(8)))  __bf16 v8bf;
typedef __attribute__((ext_vector_type(4)))  __bf16 v4bf;
typedef __attribute__((ext_vector_type(8)))  float  v8f;

// ---- LDS layout (bytes), A/B staging double-buffered ----
constexpr int A_BYTES = 32 * ASTR * 2;             // 2560
constexpr int B_BYTES = 32 * BSTR * 2;             // 8704
constexpr int SM_OUT  = 0;                         // float [MT][NT]         65536
constexpr int SM_A0   = SM_OUT  + MT * NT * 4;
constexpr int SM_A1   = SM_A0   + A_BYTES;
constexpr int SM_B0   = SM_A1   + A_BYTES;
constexpr int SM_B1   = SM_B0   + B_BYTES;
constexpr int SM_LTOK = SM_B1   + B_BYTES;         // short  [NEXP][MT]       2048
constexpr int SM_LW   = SM_LTOK + NEXP * MT * 2;   // float  [NEXP][MT]       4096
constexpr int SM_CNT  = SM_LW   + NEXP * MT * 4;   // int    [NEXP] (pad 32)    32
constexpr int SM_SE   = SM_CNT  + 32;              // int    [MT][2]          1024
constexpr int SM_SW   = SM_SE   + MT * 2 * 4;      // float  [MT][2]          1024
constexpr int SM_SIZE = SM_SW   + MT * 2 * 4;      // ~94 KB

// LDS-only barrier: waits DScnt (our LDS traffic) then split-barrier, WITHOUT
// the workgroup fence __syncthreads() implies (which forces loadcnt==0 and
// would stall the register-staged global prefetch pipeline every K-step).
__device__ __forceinline__ void lds_barrier() {
  asm volatile("s_wait_dscnt 0x0\n\t"
               "s_barrier_signal -1\n\t"
               "s_barrier_wait -1" ::: "memory");
}

__device__ __forceinline__ v4bf cvt4_bf16(float4 v, float m) {
  v4bf r;
  r[0] = (__bf16)(v.x * m);
  r[1] = (__bf16)(v.y * m);
  r[2] = (__bf16)(v.z * m);
  r[3] = (__bf16)(v.w * m);
  return r;
}

// ============================================================================
// Kernel 1: router. One wave32 per token. Computes logits = x @ gate_w + b,
// writes logits, and writes top-2 (expert ids, renormalized weights) records.
// ============================================================================
__global__ void moe_router_kernel(const float* __restrict__ x,
                                  const float* __restrict__ gate_w,
                                  const float* __restrict__ gate_b,
                                  float* __restrict__ logits_out,
                                  int*   __restrict__ rec_e,
                                  float* __restrict__ rec_w) {
  const int lane = threadIdx.x & 31;
  const int wid  = threadIdx.x >> 5;
  const int t    = blockIdx.x * 8 + wid;       // 8 waves per block, 1 token each

  const float* xp = x + (size_t)t * HSZ;

  float s[NEXP];
#pragma unroll
  for (int e = 0; e < NEXP; ++e) s[e] = 0.f;

  // lanes stripe H in float4 chunks; gate_w rows are 8 contiguous floats
#pragma unroll 2
  for (int i = 0; i < HSZ / 128; ++i) {
    const int h = i * 128 + lane * 4;
    const float4 xv = *(const float4*)(xp + h);
    const float xa[4] = { xv.x, xv.y, xv.z, xv.w };
#pragma unroll
    for (int c = 0; c < 4; ++c) {
      const float4* g = (const float4*)(gate_w + (size_t)(h + c) * NEXP);
      const float4 g0 = g[0];
      const float4 g1 = g[1];
      s[0] += xa[c] * g0.x; s[1] += xa[c] * g0.y;
      s[2] += xa[c] * g0.z; s[3] += xa[c] * g0.w;
      s[4] += xa[c] * g1.x; s[5] += xa[c] * g1.y;
      s[6] += xa[c] * g1.z; s[7] += xa[c] * g1.w;
    }
  }

  // wave32 butterfly reduction -> every lane holds all 8 sums
#pragma unroll
  for (int m = 16; m >= 1; m >>= 1) {
#pragma unroll
    for (int e = 0; e < NEXP; ++e) s[e] += __shfl_xor(s[e], m, 32);
  }

  float l[NEXP];
#pragma unroll
  for (int e = 0; e < NEXP; ++e) l[e] = s[e] + gate_b[e];

  if (lane < NEXP) logits_out[(size_t)t * NEXP + lane] = l[lane];

  if (lane == 0) {
    // top-2 by logits (softmax is monotone); ties pick lowest index like jax.top_k
    int e0 = 0; float l0 = l[0];
#pragma unroll
    for (int e = 1; e < NEXP; ++e) { if (l[e] > l0) { l0 = l[e]; e0 = e; } }
    int e1 = -1; float l1 = -3.4e38f;
#pragma unroll
    for (int e = 0; e < NEXP; ++e) { if (e != e0 && l[e] > l1) { l1 = l[e]; e1 = e; } }
    // renormalized over top-2: w0 = p0/(p0+p1) = 1/(1+exp(l1-l0))
    const float w0 = 1.0f / (1.0f + __expf(l1 - l0));
    const float w1 = 1.0f - w0;
    rec_e[t * 2 + 0] = e0;
    rec_e[t * 2 + 1] = e1;
    rec_w[t * 2 + 0] = w0;
    rec_w[t * 2 + 1] = w1;
  }
}

// ============================================================================
// Kernel 2: grouped expert GEMM with top-2 gather, bf16 WMMA, fp32 accumulate.
// Block tile: MT=128 tokens x NT=128 cols. 8 waves: 2 (M) x 4 (N), each wave
// owns a 16-row x 32-col patch (two 16x16 wmma tiles) per 32-row pass.
// Double-buffered LDS staging; one LDS-only barrier per K-step; global loads
// are register-staged two K-steps ahead.
// ============================================================================
__global__ void moe_gemm_kernel(const float* __restrict__ x,
                                const float* __restrict__ expert_w,
                                const float* __restrict__ expert_b,
                                const int*   __restrict__ rec_e,
                                const float* __restrict__ rec_w,
                                float* __restrict__ out) {
  extern __shared__ char smem[];
  float*  out_acc = (float*)(smem + SM_OUT);
  __bf16* AshA    = (__bf16*)(smem + SM_A0);
  __bf16* AshB    = (__bf16*)(smem + SM_A1);
  __bf16* BshA    = (__bf16*)(smem + SM_B0);
  __bf16* BshB    = (__bf16*)(smem + SM_B1);
  short*  ltok    = (short*)(smem + SM_LTOK);
  float*  lw      = (float*)(smem + SM_LW);
  int*    cnt     = (int*)(smem + SM_CNT);
  int*    se      = (int*)(smem + SM_SE);
  float*  sw      = (float*)(smem + SM_SW);

  const int tid  = threadIdx.x;
  const int lane = tid & 31;
  const int wid  = tid >> 5;
  const int wm   = wid >> 2;         // 0..1 : M half of the 32-row pass
  const int wn   = wid & 3;          // 0..3 : 32-col slice of NT
  const int t0   = blockIdx.x * MT;  // token tile base
  const int n0   = blockIdx.y * NT;  // column tile base

  if (tid < NEXP) cnt[tid] = 0;
  if (tid < MT) {
    se[tid * 2 + 0] = rec_e[(size_t)(t0 + tid) * 2 + 0];
    se[tid * 2 + 1] = rec_e[(size_t)(t0 + tid) * 2 + 1];
    sw[tid * 2 + 0] = rec_w[(size_t)(t0 + tid) * 2 + 0];
    sw[tid * 2 + 1] = rec_w[(size_t)(t0 + tid) * 2 + 1];
  }
  __syncthreads();

  // Build per-expert token lists. List order is non-deterministic but output
  // values are order-invariant (unique tokens per list; fixed K-loop order).
  if (tid < MT) {
#pragma unroll
    for (int k = 0; k < TOPK; ++k) {
      const int e = se[tid * 2 + k];
      const int p = atomicAdd(&cnt[e], 1);
      ltok[e * MT + p] = (short)tid;
      lw[e * MT + p]   = sw[tid * 2 + k];
    }
  }
  __syncthreads();

  // Init output accumulator with combined bias: sum_k w_k * b[e_k][n] (float4)
  for (int c = tid; c < MT * NT / 4; c += 256) {
    const int t = c >> 5;            // NT/4 = 32 chunks per token row
    const int n = (c & 31) << 2;
    const float  w0 = sw[t * 2 + 0], w1 = sw[t * 2 + 1];
    const float4 b0 = *(const float4*)&expert_b[(size_t)se[t * 2 + 0] * HSZ + n0 + n];
    const float4 b1 = *(const float4*)&expert_b[(size_t)se[t * 2 + 1] * HSZ + n0 + n];
    float4 r;
    r.x = w0 * b0.x + w1 * b1.x;  r.y = w0 * b0.y + w1 * b1.y;
    r.z = w0 * b0.z + w1 * b1.z;  r.w = w0 * b0.w + w1 * b1.w;
    *(float4*)&out_acc[t * NT + n] = r;
  }
  __syncthreads();

  // ---- per-thread staging geometry (fixed for whole kernel) ----
  // A tile 32x32: thread -> row tid/8, cols (tid%8)*4 .. +3
  const int arow_l = tid >> 3;
  const int acol   = (tid & 7) << 2;
  // B tile 32x128: thread -> rows wid+8j (j=0..3), cols lane*4 .. +3
  const int bcol   = lane << 2;

  // ---- wmma fragment geometry ----
  const int row16 = lane & 15;
  const int koff  = (lane < 16) ? 0 : 8;   // 16-bit A-matrix lane layout

  for (int e = 0; e < NEXP; ++e) {
    const int me = cnt[e];
    const float* bbase = expert_w + (size_t)e * HSZ * HSZ + n0
                       + (size_t)wid * HSZ + bcol;

    for (int rb = 0; rb < me; rb += 32) {   // 32-row pass
      v8f acc0 = {};
      v8f acc1 = {};

      // resolve this thread's gather row once per pass (clamped + masked)
      const int arow  = rb + arow_l;
      const int clamp = (arow < me) ? arow : (me - 1);
      const float amask = (arow < me) ? 1.f : 0.f;
      const float* aptr = x + (size_t)(t0 + ltok[e * MT + clamp]) * HSZ + acol;

      // ---- pipeline prologue: fill buffer 0 (k=0), prefetch k=KT ----
      {
        float4 a_reg  = *(const float4*)(aptr);
        float4 b_reg0 = *(const float4*)(bbase);
        float4 b_reg1 = *(const float4*)(bbase + (size_t)8  * HSZ);
        float4 b_reg2 = *(const float4*)(bbase + (size_t)16 * HSZ);
        float4 b_reg3 = *(const float4*)(bbase + (size_t)24 * HSZ);
        *(v4bf*)&AshA[arow_l * ASTR + acol]     = cvt4_bf16(a_reg, amask);
        *(v4bf*)&BshA[(wid + 0)  * BSTR + bcol] = cvt4_bf16(b_reg0, 1.f);
        *(v4bf*)&BshA[(wid + 8)  * BSTR + bcol] = cvt4_bf16(b_reg1, 1.f);
        *(v4bf*)&BshA[(wid + 16) * BSTR + bcol] = cvt4_bf16(b_reg2, 1.f);
        *(v4bf*)&BshA[(wid + 24) * BSTR + bcol] = cvt4_bf16(b_reg3, 1.f);
      }
      // prefetch k=KT into registers
      float4 a_reg  = *(const float4*)(aptr + KT);
      float4 b_reg0 = *(const float4*)(bbase + (size_t)(KT + 0)  * HSZ);
      float4 b_reg1 = *(const float4*)(bbase + (size_t)(KT + 8)  * HSZ);
      float4 b_reg2 = *(const float4*)(bbase + (size_t)(KT + 16) * HSZ);
      float4 b_reg3 = *(const float4*)(bbase + (size_t)(KT + 24) * HSZ);
      lds_barrier();

      const __bf16* Ar = AshA;  const __bf16* Br = BshA;   // read buffers
      __bf16*       Aw = AshB;  __bf16*       Bw = BshB;   // write buffers

      for (int k0 = 0; k0 < HSZ; k0 += KT) {
        // ---- A fragment (16x32 bf16): lane<16 holds K 0..7 & 16..23 ----
        const __bf16* Arow = Ar + (wm * 16 + row16) * ASTR;
        const v8bf alo = *(const v8bf*)(Arow + koff);
        const v8bf ahi = *(const v8bf*)(Arow + 16 + koff);
        v16bf a;
#pragma unroll
        for (int i = 0; i < 8; ++i) { a[i] = alo[i]; a[8 + i] = ahi[i]; }

        // ---- B fragments (32x16 bf16): lane = N col, K in same order as A ----
        const __bf16* Bcol = Br + wn * 32 + row16;
        v16bf b0, b1;
#pragma unroll
        for (int i = 0; i < 8; ++i) {
          b0[i] = Bcol[(koff + i) * BSTR]; b0[8 + i] = Bcol[(16 + koff + i) * BSTR];
          b1[i] = Bcol[(koff + i) * BSTR + 16]; b1[8 + i] = Bcol[(16 + koff + i) * BSTR + 16];
        }

        acc0 = __builtin_amdgcn_wmma_f32_16x16x32_bf16(false, a, false, b0,
                                                       (short)0, acc0, false, false);
        acc1 = __builtin_amdgcn_wmma_f32_16x16x32_bf16(false, a, false, b1,
                                                       (short)0, acc1, false, false);

        // ---- stage next K-step into the other buffer; prefetch k0+2*KT ----
        const int kn = k0 + KT;
        if (kn < HSZ) {
          *(v4bf*)&Aw[arow_l * ASTR + acol]     = cvt4_bf16(a_reg, amask);
          *(v4bf*)&Bw[(wid + 0)  * BSTR + bcol] = cvt4_bf16(b_reg0, 1.f);
          *(v4bf*)&Bw[(wid + 8)  * BSTR + bcol] = cvt4_bf16(b_reg1, 1.f);
          *(v4bf*)&Bw[(wid + 16) * BSTR + bcol] = cvt4_bf16(b_reg2, 1.f);
          *(v4bf*)&Bw[(wid + 24) * BSTR + bcol] = cvt4_bf16(b_reg3, 1.f);
          const int kp = k0 + 2 * KT;
          if (kp < HSZ) {
            const float* bk = bbase + (size_t)kp * HSZ;
            a_reg  = *(const float4*)(aptr + kp);
            b_reg0 = *(const float4*)(bk);
            b_reg1 = *(const float4*)(bk + (size_t)8  * HSZ);
            b_reg2 = *(const float4*)(bk + (size_t)16 * HSZ);
            b_reg3 = *(const float4*)(bk + (size_t)24 * HSZ);
          }
        }

        lds_barrier();   // one LDS-only barrier per K-step (double-buffered)
        const __bf16* tA = Ar; Ar = Aw; Aw = (__bf16*)tA;
        const __bf16* tB = Br; Br = Bw; Bw = (__bf16*)tB;
      }

      // ---- weighted scatter-accumulate into LDS output tile ----
      // C layout: VGPR r -> (M = r + 8*(lane>=16), N = lane&15)
      const int nlo  = wn * 32 + (lane & 15);
      const int mofs = wm * 16 + ((lane >> 4) << 3);
#pragma unroll
      for (int r = 0; r < 8; ++r) {
        const int mrow = rb + mofs + r;
        if (mrow < me) {
          const int tok = ltok[e * MT + mrow];
          const float w = lw[e * MT + mrow];
          out_acc[tok * NT + nlo]      += w * acc0[r];
          out_acc[tok * NT + nlo + 16] += w * acc1[r];
        }
      }
      lds_barrier();   // order passes: a token's e0/e1 rows may hit same addr
    }
  }

  __syncthreads();
  // ---- store output tile, float4 coalesced ----
  for (int c = tid; c < MT * NT / 4; c += 256) {
    const int t = c >> 5;
    const int n = (c & 31) << 2;
    *(float4*)&out[(size_t)(t0 + t) * HSZ + n0 + n] = *(const float4*)&out_acc[t * NT + n];
  }
}

// ============================================================================
extern "C" void kernel_launch(void* const* d_in, const int* in_sizes, int n_in,
                              void* d_out, int out_size, void* d_ws, size_t ws_size,
                              hipStream_t stream) {
  (void)in_sizes; (void)n_in; (void)out_size; (void)ws_size;

  const float* x        = (const float*)d_in[0];   // [T, H]
  const float* gate_w   = (const float*)d_in[1];   // [H, E]
  const float* gate_b   = (const float*)d_in[2];   // [E]
  const float* expert_w = (const float*)d_in[3];   // [E, H, H]
  const float* expert_b = (const float*)d_in[4];   // [E, H]

  float* out    = (float*)d_out;                   // [T, H]
  float* logits = out + (size_t)TOK * HSZ;         // [T, E]

  // scratch: per-token top-2 records (2*T ints + 2*T floats = 256 KB)
  int*   rec_e = (int*)d_ws;
  float* rec_w = (float*)d_ws + (size_t)2 * TOK;

  moe_router_kernel<<<TOK / 8, 256, 0, stream>>>(x, gate_w, gate_b,
                                                 logits, rec_e, rec_w);

  dim3 grid(TOK / MT, HSZ / NT);                   // 128 x 8
  moe_gemm_kernel<<<grid, 256, SM_SIZE, stream>>>(x, expert_w, expert_b,
                                                  rec_e, rec_w, out);
}
